// RNN_44006234914818
// MI455X (gfx1250) — matrix operand order repaired and checked by reading, not compiled, not run on hardware
//
#include <hip/hip_runtime.h>

#define B_   64
#define T_   512
#define DK   512           // D == A_DIM == U == 512
#define NWG_SCAN 16

#define AS_G 40            // padded LDS row stride (ushorts) for GEMM tiles
#define WS_S 520           // padded LDS row stride (ushorts) for scan arrays

typedef __attribute__((ext_vector_type(16))) __bf16 v16bf;
typedef __attribute__((ext_vector_type(8)))  float  v8f;
typedef __attribute__((ext_vector_type(4)))  unsigned int u32x4;
typedef __attribute__((ext_vector_type(8)))  int i32x8;
typedef __attribute__((ext_vector_type(4)))  int i32x4;

union ABFrag { unsigned int u[8]; v16bf v; };

#if defined(__has_builtin)
#if __has_builtin(__builtin_amdgcn_tensor_load_to_lds) && __has_builtin(__builtin_amdgcn_s_wait_tensorcnt)
#define USE_TDM 1
#endif
#endif

__device__ __forceinline__ unsigned short f2bf(float f) {
    unsigned int u = __builtin_bit_cast(unsigned int, f);
    u += 0x7FFFu + ((u >> 16) & 1u);          // round-to-nearest-even
    return (unsigned short)(u >> 16);
}
__device__ __forceinline__ unsigned int pack_bf16(float lo, float hi) {
    return (unsigned int)f2bf(lo) | ((unsigned int)f2bf(hi) << 16);
}
// branchless tanh: 1 - 2/(e^{2x}+1), via v_exp_f32 (2^x) and v_rcp_f32.
// |err| ~1e-6, far below bf16 storage precision; clamp keeps exp2 finite.
__device__ __forceinline__ float fast_tanh(float x) {
    float xc = fminf(fmaxf(x, -9.0f), 9.0f);
    float e  = __builtin_amdgcn_exp2f(xc * 2.885390081777927f);  // e^{2x}
    return 1.0f - 2.0f * __builtin_amdgcn_rcpf(e + 1.0f);
}

// ---------------- init: convert weights to bf16, zero state + barrier ----------------
__global__ void init_convert(const float* __restrict__ Waa,
                             const float* __restrict__ Wax,
                             const float* __restrict__ Wya,
                             unsigned short* __restrict__ Waa_bf,
                             unsigned short* __restrict__ Wax_bf,
                             unsigned short* __restrict__ Wya_bf,
                             unsigned int* __restrict__ Abuf0_u32,
                             int* __restrict__ bar) {
    int tid = blockIdx.x * blockDim.x + threadIdx.x;
    if (tid < DK * DK) {
        Waa_bf[tid] = f2bf(Waa[tid]);
        Wax_bf[tid] = f2bf(Wax[tid]);
        Wya_bf[tid] = f2bf(Wya[tid]);
    }
    if (tid < (B_ * DK) / 2) Abuf0_u32[tid] = 0u;   // a0 = 0
    if (tid < 2) bar[tid] = 0;                       // barrier count / generation
}

// ---------------- batched GEMM: out[Mx512] = A[Mx512] * W_bf16[512x512] + bias --------
// ABF=true: A is bf16 (Abf); ABF=false: A is f32 (Af32), converted while staging.
// swz=1: output row m remapped to (m % T_)*B_ + m/T_  (stores XW as [T][B][A])
template <bool ABF>
__global__ void __launch_bounds__(256)
gemm_bf16(const float* __restrict__ Af32,
          const unsigned short* __restrict__ Abf,
          const unsigned short* __restrict__ W,
          const float* __restrict__ bias,
          float* __restrict__ out,
          int nblk_n, int swz) {
    __shared__ __align__(16) unsigned short Alds[64 * AS_G];   // [r][k] padded
    __shared__ __align__(16) unsigned short Blds[32 * AS_G];   // [n][k] transposed, padded

    const int bm = blockIdx.x / nblk_n;
    const int bn = blockIdx.x % nblk_n;
    const int m0 = bm * 64;
    const int n0 = bn * 32;

    const int tid  = threadIdx.x;
    const int wave = tid >> 5;
    const int lane = tid & 31;
    const int mt   = wave & 3;     // 4 M-tiles
    const int nt   = wave >> 2;    // 2 N-tiles
    const int l15  = lane & 15;
    const int hi   = lane >> 4;

    // staging coordinates (fixed per thread)
    const int ar  = tid >> 2;            // 0..63  A row
    const int ak0 = (tid & 3) * 8;       // 0,8,16,24  8 consecutive k
    const int bk  = tid >> 3;            // 0..31  W row (k)
    const int bn0 = (tid & 7) * 4;       // 4 consecutive n

    v8f acc = {};

    for (int kc = 0; kc < DK / 32; ++kc) {
        __syncthreads();
        // ---- stage A tile 64x32, one ds_store_b128 per thread ----
        if (ABF) {
            *(uint4*)&Alds[ar * AS_G + ak0] =
                *(const uint4*)(Abf + (m0 + ar) * DK + kc * 32 + ak0);
        } else {
            const float4* s = (const float4*)(Af32 + (m0 + ar) * DK + kc * 32 + ak0);
            float4 x0 = s[0], x1 = s[1];
            uint4 p;
            p.x = pack_bf16(x0.x, x0.y);  p.y = pack_bf16(x0.z, x0.w);
            p.z = pack_bf16(x1.x, x1.y);  p.w = pack_bf16(x1.z, x1.w);
            *(uint4*)&Alds[ar * AS_G + ak0] = p;
        }
        // ---- stage B tile 32x32 transposed into [n][k] ----
        {
            uint2 w2 = *(const uint2*)(W + (kc * 32 + bk) * DK + n0 + bn0);
            Blds[(bn0 + 0) * AS_G + bk] = (unsigned short)(w2.x);
            Blds[(bn0 + 1) * AS_G + bk] = (unsigned short)(w2.x >> 16);
            Blds[(bn0 + 2) * AS_G + bk] = (unsigned short)(w2.y);
            Blds[(bn0 + 3) * AS_G + bk] = (unsigned short)(w2.y >> 16);
        }
        __syncthreads();

        // A fragment: lane holds row (mt*16+l15); K pairs per ISA 7.12.2 layout
        ABFrag a, b;
        const int r   = mt * 16 + l15;
        const int koA = hi ? 8 : 0;
        #pragma unroll
        for (int i = 0; i < 8; ++i) {
            int keven = koA + ((i & 4) ? 16 : 0) + 2 * (i & 3);
            a.u[i] = *(const unsigned int*)&Alds[r * AS_G + keven];
        }
        // B fragment: lane holds column n, K pairs in same layout as A
        const int n   = nt * 16 + l15;
        const int koB = hi ? 8 : 0;
        #pragma unroll
        for (int i = 0; i < 8; ++i)
            b.u[i] = *(const unsigned int*)&Blds[n * AS_G + koB + ((i & 4) ? 16 : 0) + 2 * (i & 3)];

        acc = __builtin_amdgcn_wmma_f32_16x16x32_bf16(false, a.v, false, b.v,
                                                      (short)0, acc, false, false);
    }

    const int   n_g  = n0 + nt * 16 + l15;
    const float bv   = bias[n_g];
    const int   mloc = m0 + mt * 16 + hi * 8;
    #pragma unroll
    for (int v = 0; v < 8; ++v) {
        int m    = mloc + v;
        int orow = swz ? ((m & (T_ - 1)) * B_ + (m >> 9)) : m;
        out[orow * DK + n_g] = acc[v] + bv;
    }
}

// ---------------- serial recurrence: a <- tanh(a @ Waa + XW[t]) ----------------
// LDS layout (dynamic): Wlds [32][WS_S] at byte 0 (33280 B),
//                       Alds [64][WS_S] at byte 33280 (66560 B)
#define LDS_A_OFF 33280u

__global__ void __launch_bounds__(256)
rnn_scan(const unsigned short* __restrict__ Waa_bf,
         const float* __restrict__ XW,            // [T][B][A] f32 (lives in d_out)
         unsigned short* __restrict__ Abuf0,      // ping
         unsigned short* __restrict__ Abuf1,      // pong
         unsigned short* __restrict__ Aseq,       // [B][T][A] bf16
         int* __restrict__ bar) {
    extern __shared__ __align__(16) unsigned short smem[];
    unsigned short* Wlds = smem;                       // [32][WS_S]
    unsigned short* Alds = smem + 32 * WS_S;           // [64][WS_S]

    const int wg   = blockIdx.x;              // 0..15, owns 32 output columns
    const int n0   = wg * 32;
    const int tid  = threadIdx.x;
    const int wave = tid >> 5;
    const int lane = tid & 31;
    const int mt   = wave & 3;
    const int nt   = wave >> 2;
    const int l15  = lane & 15;
    const int hi   = lane >> 4;

    // stage this WG's Waa column slice once: Wlds[n][k], padded rows
    for (int idx = tid; idx < 32 * DK; idx += 256) {
        int nn = idx & 31, kk = idx >> 5;
        Wlds[nn * WS_S + kk] = Waa_bf[kk * DK + n0 + nn];
    }
    __syncthreads();

    const int n_loc = nt * 16 + l15;
    const int n_g   = n0 + n_loc;
    const int mb    = mt * 16 + hi * 8;
    const int koA   = hi ? 8 : 0;
    const int koB   = hi ? 8 : 0;
    const int arow  = (mt * 16 + l15) * WS_S;
    const int brow  = n_loc * WS_S;

    // B fragments are invariant across all 512 steps: hoist into registers
    // (16 chunks x 8 VGPRs = 128 VGPRs; only 8 waves/WGP so pressure is free)
    ABFrag bfr[16];
    #pragma unroll
    for (int kc = 0; kc < 16; ++kc)
        #pragma unroll
        for (int i = 0; i < 8; ++i)
            bfr[kc].u[i] = *(const unsigned int*)&Wlds[brow + kc * 32 + koB + ((i & 4) ? 16 : 0) + 2 * (i & 3)];

    const unsigned short* cur = Abuf0;
    unsigned short*       nxt = Abuf1;

    // preload step-0 pre-activations into registers
    float xw[8];
    #pragma unroll
    for (int v = 0; v < 8; ++v)
        xw[v] = XW[(0 * B_ + mb + v) * DK + n_g];

    for (int t = 0; t < T_; ++t) {
        __syncthreads();
#ifdef USE_TDM
        // Tensor Data Mover: DMA the 64x512 bf16 hidden state into LDS with
        // pad_enable inserting 4 dwords every 256 dwords -> 520-ushort rows.
        if (wave == 0) {
            unsigned long long ga = (unsigned long long)(const void*)cur;
            u32x4 g0;
            g0[0] = 1u;                                        // count=1, user mode
            g0[1] = LDS_A_OFF;                                 // lds_addr (bytes)
            g0[2] = (unsigned int)(ga & 0xFFFFFFFFull);        // global_addr[31:0]
            g0[3] = (unsigned int)((ga >> 32) & 0x01FFFFFFull) // global_addr[56:32]
                    | 0x80000000u;                             // type=2 (image)
            i32x8 g1;
            g1[0] = (int)0x07D10000;   // mask=0 | ds=1<<16 | pad_en | intv=7 | amt=3
            g1[1] = (int)0x80000000;   // tensor_dim0 = 32768 (low16 in bits 63:48)
            g1[2] = (int)0x00010000;   // tensor_dim0 hi=0, tensor_dim1 = 1
            g1[3] = (int)0x80000000;   // tile_dim0 = 32768 (bits 127:112)
            g1[4] = 0;                 // tile_dim1=0, tile_dim2=0
            g1[5] = 32768;             // tensor_dim0_stride
            g1[6] = 0; g1[7] = 0;
            i32x4 gz = {0, 0, 0, 0};
#if __clang_major__ >= 23
            i32x8 gz8 = {0, 0, 0, 0, 0, 0, 0, 0};
            __builtin_amdgcn_tensor_load_to_lds(g0, g1, gz, gz, gz8, 0);
#else
            __builtin_amdgcn_tensor_load_to_lds(g0, g1, gz, gz, 0);
#endif
            __builtin_amdgcn_s_wait_tensorcnt(0);
        }
#else
        // fallback: cooperative 128-bit copies into padded rows
        #pragma unroll
        for (int j = 0; j < 16; ++j) {
            int c = tid + j * 256;             // 4096 uint4 chunks
            int r = c >> 6, cc = c & 63;
            *((uint4*)(Alds + r * WS_S) + cc) = *((const uint4*)(cur + r * DK) + cc);
        }
#endif
        __syncthreads();

        v8f acc = {};
        #pragma unroll
        for (int kc = 0; kc < 16; ++kc) {
            ABFrag a;
            #pragma unroll
            for (int i = 0; i < 8; ++i) {
                int keven = koA + ((i & 4) ? 16 : 0) + 2 * (i & 3);
                a.u[i] = *(const unsigned int*)&Alds[arow + kc * 32 + keven];
            }
            acc = __builtin_amdgcn_wmma_f32_16x16x32_bf16(false, a.v, false, bfr[kc].v,
                                                          (short)0, acc, false, false);
        }

        // consume the preloaded pre-activations
        float pre[8];
        #pragma unroll
        for (int v = 0; v < 8; ++v) pre[v] = acc[v] + xw[v];

        // preload next step's XW (read-only) so load latency hides under the
        // barrier + TDM + WMMA of the next iteration
        const int tn = (t + 1 < T_) ? t + 1 : t;
        #pragma unroll
        for (int v = 0; v < 8; ++v)
            xw[v] = XW[(tn * B_ + mb + v) * DK + n_g];
        if (t + 2 < T_) {
            const char* p = (const char*)&XW[(size_t)(t + 2) * B_ * DK];
            __builtin_prefetch(p + tid * 512, 0, 1);
        }

        // tanh + bf16 stores (ping-pong state + sequence for phase 3)
        #pragma unroll
        for (int v = 0; v < 8; ++v) {
            int b = mb + v;
            unsigned short hb = f2bf(fast_tanh(pre[v]));
            nxt[b * DK + n_g]             = hb;
            Aseq[(b * T_ + t) * DK + n_g] = hb;
        }

        // device-wide barrier across the 16 persistent workgroups
        __threadfence();
        __syncthreads();
        if (tid == 0) {
            volatile int* gen = bar + 1;
            int g = *gen;
            if (atomicAdd(bar, 1) == NWG_SCAN - 1) {
                bar[0] = 0;
                __threadfence();
                atomicAdd((int*)gen, 1);
            } else {
                while (*gen == g) __builtin_amdgcn_s_sleep(1);
            }
        }
        __syncthreads();

        const unsigned short* tmp = nxt;
        nxt = (unsigned short*)cur;
        cur = tmp;
    }
}

extern "C" void kernel_launch(void* const* d_in, const int* in_sizes, int n_in,
                              void* d_out, int out_size, void* d_ws, size_t ws_size,
                              hipStream_t stream) {
    (void)in_sizes; (void)n_in; (void)out_size; (void)ws_size;
    const float* X   = (const float*)d_in[0];
    const float* Waa = (const float*)d_in[1];
    const float* Wax = (const float*)d_in[2];
    const float* ba  = (const float*)d_in[3];
    const float* Wya = (const float*)d_in[4];
    const float* by  = (const float*)d_in[5];
    float* out = (float*)d_out;

    char* ws = (char*)d_ws;
    unsigned short* Wax_bf = (unsigned short*)(ws + (size_t)0);
    unsigned short* Waa_bf = (unsigned short*)(ws + ((size_t)512 << 10));
    unsigned short* Wya_bf = (unsigned short*)(ws + ((size_t)1024 << 10));
    unsigned short* Abuf0  = (unsigned short*)(ws + ((size_t)1536 << 10));
    unsigned short* Abuf1  = (unsigned short*)(ws + ((size_t)1600 << 10));
    int*            bar    = (int*)          (ws + ((size_t)1664 << 10));
    unsigned short* Aseq   = (unsigned short*)(ws + ((size_t)2048 << 10));

    // Phase 0: weight conversion + state/barrier init
    init_convert<<<1024, 256, 0, stream>>>(Waa, Wax, Wya, Waa_bf, Wax_bf, Wya_bf,
                                           (unsigned int*)Abuf0, bar);

    // Phase 1: XW = X @ Wax + ba, stored in d_out as [T][B][A]
    const int nblk_n = DK / 32;                      // 16
    const int nblk   = (B_ * T_ / 64) * nblk_n;      // 8192
    gemm_bf16<false><<<nblk, 256, 0, stream>>>(X, nullptr, Wax_bf, ba, out,
                                               nblk_n, /*swz=*/1);

    // Phase 2: serial scan, 16 persistent WGs, padded LDS (Wlds+Alds = 99840 B)
    rnn_scan<<<NWG_SCAN, 256, (32 + 64) * WS_S * 2, stream>>>(
        Waa_bf, out, Abuf0, Abuf1, Aseq, bar);

    // Phase 3: Y = Aseq @ Wya + by, overwrites d_out as [B][T][U]
    gemm_bf16<true><<<nblk, 256, 0, stream>>>(nullptr, Aseq, Wya_bf, by, out,
                                              nblk_n, /*swz=*/0);
}
